// MoELayer_86002425135559
// MI455X (gfx1250) — matrix operand order repaired and checked by reading, not compiled, not run on hardware
//
#include <hip/hip_runtime.h>
#include <hip/hip_bf16.h>
#include <math.h>

typedef __attribute__((ext_vector_type(16))) __bf16 v16bf;
typedef __attribute__((ext_vector_type(8)))  __bf16 bf8;
typedef __attribute__((ext_vector_type(4)))  __bf16 bf4;
typedef __attribute__((ext_vector_type(8)))  float  v8f;

static constexpr int Bb   = 4;
static constexpr int Ss   = 2048;
static constexpr int Hd   = 1024;
static constexpr int Ee   = 8;
static constexpr int Fd   = 4096;
static constexpr int Ntok = Bb * Ss;          // 8192

static constexpr int TM  = 64;                // token tile
static constexpr int TN  = 128;               // output-feature tile (8 n-subtiles)
static constexpr int TK  = 32;                // K step (bf16 WMMA K)
static constexpr int LDX = 40;                // A row stride: 80B, 16B-aligned, 20-bank stride
static constexpr int LDB = 24;                // B lane stride: 48B, 16B-aligned, 12-bank stride

__device__ __forceinline__ int imin(int a, int b) { return a < b ? a : b; }

// tanh-approx gelu in sigmoid form: 0.5x(1+tanh(u)) == x * sigmoid(2u); branch-free.
__device__ __forceinline__ float gelu_f(float x) {
  float u2 = 1.5957691216057308f * (x + 0.044715f * x * x * x);  // 2*sqrt(2/pi)*(x+cx^3)
  return x / (1.0f + __expf(-u2));
}

// A fragment (16x32 bf16): lane row = r0+(lane&15); lanes>=16 start at K=8.
// Per lane: K = kb..kb+7 and kb+16..kb+23 -> two contiguous 16B segments.
__device__ __forceinline__ v16bf frag_a(const __bf16* Xs, int r0, int lane) {
  const __bf16* p = Xs + (r0 + (lane & 15)) * LDX + ((lane >> 4) << 3);
  bf8 lo = *(const bf8*)(p);
  bf8 hi = *(const bf8*)(p + 16);
  return __builtin_shufflevector(lo, hi, 0, 1, 2, 3, 4, 5, 6, 7, 8, 9, 10, 11, 12, 13, 14, 15);
}

// B fragment: stored in fragment order BF[tn][lane][j]; lane's 16 j contiguous.
__device__ __forceinline__ v16bf frag_b(const __bf16* BF, int tn, int lane) {
  const __bf16* p = BF + tn * (32 * LDB) + lane * LDB;
  bf8 lo = *(const bf8*)(p);
  bf8 hi = *(const bf8*)(p + 8);
  return __builtin_shufflevector(lo, hi, 0, 1, 2, 3, 4, 5, 6, 7, 8, 9, 10, 11, 12, 13, 14, 15);
}

// ---------------- router: one wave per token ----------------
__global__ __launch_bounds__(256) void moe_router(
    const float* __restrict__ hs, const float* __restrict__ wr,
    float* __restrict__ probs, int* __restrict__ expert_of,
    float* __restrict__ gates, int* __restrict__ counts) {
  const int lane = threadIdx.x & 31;
  const int t = blockIdx.x * 8 + (threadIdx.x >> 5);
  const float* x = hs + (size_t)t * Hd;
  float acc[Ee];
#pragma unroll
  for (int e = 0; e < Ee; ++e) acc[e] = 0.0f;
  for (int h = lane; h < Hd; h += 32) {
    float xv = x[h];
#pragma unroll
    for (int e = 0; e < Ee; ++e) acc[e] += xv * wr[h * Ee + e];
  }
#pragma unroll
  for (int e = 0; e < Ee; ++e) {
    float v = acc[e];
#pragma unroll
    for (int off = 16; off > 0; off >>= 1) v += __shfl_xor(v, off, 32);
    acc[e] = v;
  }
  if (lane == 0) {
    float mx = acc[0];
#pragma unroll
    for (int e = 1; e < Ee; ++e) mx = fmaxf(mx, acc[e]);
    float p[Ee], s = 0.0f;
#pragma unroll
    for (int e = 0; e < Ee; ++e) { p[e] = __expf(acc[e] - mx); s += p[e]; }
    float inv = 1.0f / s;
    int arg = 0; float best = -1.0f;
#pragma unroll
    for (int e = 0; e < Ee; ++e) {
      p[e] *= inv;
      if (p[e] > best) { best = p[e]; arg = e; }   // first-max tie break
      probs[(size_t)t * Ee + e] = p[e];
    }
    expert_of[t] = arg;
    gates[t] = best;
    atomicAdd(&counts[arg], 1);
  }
}

__global__ void moe_init(int* counts) {
  if (threadIdx.x < Ee) counts[threadIdx.x] = 0;
}

__global__ void moe_scan_loss(const int* __restrict__ counts, int* __restrict__ offs,
                              int* __restrict__ cursor, float* __restrict__ loss_out) {
  if (threadIdx.x == 0 && blockIdx.x == 0) {
    int o = 0; float ss = 0.0f;
    for (int e = 0; e < Ee; ++e) {
      offs[e] = o; cursor[e] = o; o += counts[e];
      float c = (float)counts[e]; ss += c * c;
    }
    offs[Ee] = o;
    float n = (float)o;
    loss_out[0] = ss / (n * n);   // mean(c^2)*E/(sum c)^2 == sum(c^2)/N^2
  }
}

__global__ __launch_bounds__(256) void moe_scatter(
    const int* __restrict__ expert_of, int* __restrict__ cursor, int* __restrict__ perm) {
  int t = blockIdx.x * blockDim.x + threadIdx.x;
  if (t < Ntok) {
    int e = expert_of[t];
    int slot = atomicAdd(&cursor[e], 1);
    perm[slot] = t;
  }
}

// gbw[e][h] = gelu(bi[e]) @ wo[e]   (non-selected-expert constant path)
__global__ __launch_bounds__(256) void moe_gbw(
    const float* __restrict__ bi, const float* __restrict__ wo, float* __restrict__ gbw) {
  const int e = blockIdx.y;
  const int h = blockIdx.x * blockDim.x + threadIdx.x;
  const float* wo_e = wo + (size_t)e * Fd * Hd;
  const float* bi_e = bi + (size_t)e * Fd;
  float acc = 0.0f;
  for (int f = 0; f < Fd; ++f) acc += gelu_f(bi_e[f]) * wo_e[(size_t)f * Hd + h];
  gbw[e * Hd + h] = acc;
}

__global__ __launch_bounds__(256) void moe_call(
    const float* __restrict__ gbw, const float* __restrict__ bo, float* __restrict__ call) {
  const int h = blockIdx.x * blockDim.x + threadIdx.x;
  float s = 0.0f;
#pragma unroll
  for (int e = 0; e < Ee; ++e) s += gbw[e * Hd + h] + bo[e * Hd + h];
  call[h] = s;
}

// Scatter 4 converted weight values into fragment-order B tile.
__device__ __forceinline__ void stage_b4(__bf16* BF, int k, int n4, float4 wv) {
  const int j = (k & 7) | ((k >> 4) << 3);
  const int kh = ((k >> 3) & 1) << 4;
  float arr[4] = {wv.x, wv.y, wv.z, wv.w};
#pragma unroll
  for (int q = 0; q < 4; ++q) {
    int n = n4 + q;
    BF[(n >> 4) * (32 * LDB) + (((n & 15) + kh)) * LDB + j] = (__bf16)arr[q];
  }
}

// ---------------- GEMM1: h1 = gelu((x*gate) @ wi[e] + bi[e]) -> bf16 ----------------
__global__ __launch_bounds__(256) void moe_gemm1(
    const float* __restrict__ hidden, const float* __restrict__ wi,
    const float* __restrict__ bi, const float* __restrict__ gates,
    const int* __restrict__ perm, const int* __restrict__ offs,
    __bf16* __restrict__ h1) {
  __shared__ __bf16 Xs[TM * LDX];
  __shared__ __bf16 BF[8 * 32 * LDB];
  __shared__ int   toks[TM];
  __shared__ float gts[TM];

  const int e = blockIdx.z;
  const int row_start = offs[e] + blockIdx.x * TM;
  const int row_end   = offs[e + 1];
  if (row_start >= row_end) return;
  const int f0 = blockIdx.y * TN;
  const int tid = threadIdx.x, lane = tid & 31, wid = tid >> 5;
  const int wm = wid >> 2, wn = wid & 3;
  const float* wi_e = wi + (size_t)e * Hd * Fd;

  if (tid < TM) {
    int row = imin(row_start + tid, row_end - 1);   // clamp: pad rows read valid data
    int t = perm[row];
    toks[tid] = t;
    gts[tid] = gates[t];
  }

  const int ar = tid >> 3, ak = (tid & 7) << 2;           // A stage coords (2 iters)
  const int ar2 = ar + 32;
  const v8f vzero = {0.f, 0.f, 0.f, 0.f, 0.f, 0.f, 0.f, 0.f};
  v8f acc[2][2];
#pragma unroll
  for (int i = 0; i < 2; ++i)
#pragma unroll
    for (int j = 0; j < 2; ++j) acc[i][j] = vzero;

  __syncthreads();                                        // toks/gts visible

  // prologue prefetch (k0 = 0)
  float4 aP0 = *(const float4*)(hidden + (size_t)toks[ar]  * Hd + ak);
  float4 aP1 = *(const float4*)(hidden + (size_t)toks[ar2] * Hd + ak);
  float4 bP[4];
#pragma unroll
  for (int it = 0; it < 4; ++it) {
    int idx = tid + it * 256;
    int k = idx >> 5, n4 = (idx & 31) << 2;
    bP[it] = *(const float4*)(wi_e + (size_t)k * Fd + f0 + n4);
  }

  for (int k0 = 0; k0 < Hd; k0 += TK) {
    __syncthreads();                                      // prev compute done reading LDS
    {
      float g0 = gts[ar], g1 = gts[ar2];
      bf4 o0, o1;
      o0[0] = (__bf16)(aP0.x * g0); o0[1] = (__bf16)(aP0.y * g0);
      o0[2] = (__bf16)(aP0.z * g0); o0[3] = (__bf16)(aP0.w * g0);
      o1[0] = (__bf16)(aP1.x * g1); o1[1] = (__bf16)(aP1.y * g1);
      o1[2] = (__bf16)(aP1.z * g1); o1[3] = (__bf16)(aP1.w * g1);
      *(bf4*)(Xs + ar  * LDX + ak) = o0;
      *(bf4*)(Xs + ar2 * LDX + ak) = o1;
    }
#pragma unroll
    for (int it = 0; it < 4; ++it) {
      int idx = tid + it * 256;
      int k = idx >> 5, n4 = (idx & 31) << 2;
      stage_b4(BF, k, n4, bP[it]);
    }
    __syncthreads();
    if (k0 + TK < Hd) {                                   // prefetch next tile (overlaps WMMA)
      int kn = k0 + TK;
      aP0 = *(const float4*)(hidden + (size_t)toks[ar]  * Hd + kn + ak);
      aP1 = *(const float4*)(hidden + (size_t)toks[ar2] * Hd + kn + ak);
#pragma unroll
      for (int it = 0; it < 4; ++it) {
        int idx = tid + it * 256;
        int k = idx >> 5, n4 = (idx & 31) << 2;
        bP[it] = *(const float4*)(wi_e + (size_t)(kn + k) * Fd + f0 + n4);
      }
    }
    v16bf a0 = frag_a(Xs, wm * 32,      lane);
    v16bf a1 = frag_a(Xs, wm * 32 + 16, lane);
    v16bf b0 = frag_b(BF, wn * 2,     lane);
    v16bf b1 = frag_b(BF, wn * 2 + 1, lane);
    acc[0][0] = __builtin_amdgcn_wmma_f32_16x16x32_bf16(false, a0, false, b0, (short)0, acc[0][0], false, false);
    acc[0][1] = __builtin_amdgcn_wmma_f32_16x16x32_bf16(false, a0, false, b1, (short)0, acc[0][1], false, false);
    acc[1][0] = __builtin_amdgcn_wmma_f32_16x16x32_bf16(false, a1, false, b0, (short)0, acc[1][0], false, false);
    acc[1][1] = __builtin_amdgcn_wmma_f32_16x16x32_bf16(false, a1, false, b1, (short)0, acc[1][1], false, false);
  }

  const float* bi_e = bi + (size_t)e * Fd;
#pragma unroll
  for (int i = 0; i < 2; ++i)
#pragma unroll
    for (int j = 0; j < 2; ++j) {
      int rbase = wm * 32 + i * 16 + ((lane >> 4) << 3);
      int col = f0 + wn * 32 + j * 16 + (lane & 15);
#pragma unroll
      for (int v = 0; v < 8; ++v) {
        int row = row_start + rbase + v;
        if (row < row_end)
          h1[(size_t)row * Fd + col] = (__bf16)gelu_f(acc[i][j][v] + bi_e[col]);
      }
    }
}

// ---------------- GEMM2: out = h1 @ wo[e] + Call - gbw[e] ----------------
__global__ __launch_bounds__(256) void moe_gemm2(
    const __bf16* __restrict__ h1, const float* __restrict__ wo,
    const int* __restrict__ perm, const int* __restrict__ offs,
    const float* __restrict__ gbw, const float* __restrict__ call,
    float* __restrict__ out) {
  __shared__ __bf16 Xs[TM * LDX];
  __shared__ __bf16 BF[8 * 32 * LDB];
  __shared__ int toks[TM];

  const int e = blockIdx.z;
  const int row_start = offs[e] + blockIdx.x * TM;
  const int row_end   = offs[e + 1];
  if (row_start >= row_end) return;
  const int h0 = blockIdx.y * TN;
  const int tid = threadIdx.x, lane = tid & 31, wid = tid >> 5;
  const int wm = wid >> 2, wn = wid & 3;
  const float* wo_e = wo + (size_t)e * Fd * Hd;

  if (tid < TM) {
    int row = imin(row_start + tid, row_end - 1);
    toks[tid] = perm[row];
  }

  const int ar = tid >> 2, ak = (tid & 3) << 3;           // A stage coords (1 iter, bf16x8)
  const int arow = imin(row_start + ar, row_end - 1);
  const v8f vzero = {0.f, 0.f, 0.f, 0.f, 0.f, 0.f, 0.f, 0.f};
  v8f acc[2][2];
#pragma unroll
  for (int i = 0; i < 2; ++i)
#pragma unroll
    for (int j = 0; j < 2; ++j) acc[i][j] = vzero;

  __syncthreads();

  // prologue prefetch (k0 = 0)
  bf8 aP = *(const bf8*)(h1 + (size_t)arow * Fd + ak);
  float4 bP[4];
#pragma unroll
  for (int it = 0; it < 4; ++it) {
    int idx = tid + it * 256;
    int k = idx >> 5, n4 = (idx & 31) << 2;
    bP[it] = *(const float4*)(wo_e + (size_t)k * Hd + h0 + n4);
  }

  for (int k0 = 0; k0 < Fd; k0 += TK) {
    __syncthreads();
    *(bf8*)(Xs + ar * LDX + ak) = aP;
#pragma unroll
    for (int it = 0; it < 4; ++it) {
      int idx = tid + it * 256;
      int k = idx >> 5, n4 = (idx & 31) << 2;
      stage_b4(BF, k, n4, bP[it]);
    }
    __syncthreads();
    if (k0 + TK < Fd) {
      int kn = k0 + TK;
      aP = *(const bf8*)(h1 + (size_t)arow * Fd + kn + ak);
#pragma unroll
      for (int it = 0; it < 4; ++it) {
        int idx = tid + it * 256;
        int k = idx >> 5, n4 = (idx & 31) << 2;
        bP[it] = *(const float4*)(wo_e + (size_t)(kn + k) * Hd + h0 + n4);
      }
    }
    v16bf a0 = frag_a(Xs, wm * 32,      lane);
    v16bf a1 = frag_a(Xs, wm * 32 + 16, lane);
    v16bf b0 = frag_b(BF, wn * 2,     lane);
    v16bf b1 = frag_b(BF, wn * 2 + 1, lane);
    acc[0][0] = __builtin_amdgcn_wmma_f32_16x16x32_bf16(false, a0, false, b0, (short)0, acc[0][0], false, false);
    acc[0][1] = __builtin_amdgcn_wmma_f32_16x16x32_bf16(false, a0, false, b1, (short)0, acc[0][1], false, false);
    acc[1][0] = __builtin_amdgcn_wmma_f32_16x16x32_bf16(false, a1, false, b0, (short)0, acc[1][0], false, false);
    acc[1][1] = __builtin_amdgcn_wmma_f32_16x16x32_bf16(false, a1, false, b1, (short)0, acc[1][1], false, false);
  }

  const float* gbw_e = gbw + (size_t)e * Hd;
#pragma unroll
  for (int i = 0; i < 2; ++i)
#pragma unroll
    for (int j = 0; j < 2; ++j) {
      int rbase = wm * 32 + i * 16 + ((lane >> 4) << 3);
      int col = h0 + wn * 32 + j * 16 + (lane & 15);
#pragma unroll
      for (int v = 0; v < 8; ++v) {
        int row = row_start + rbase + v;
        if (row < row_end) {
          int t = toks[rbase + v];
          out[(size_t)t * Hd + col] = acc[i][j][v] + call[col] - gbw_e[col];
        }
      }
    }
}

extern "C" void kernel_launch(void* const* d_in, const int* in_sizes, int n_in,
                              void* d_out, int out_size, void* d_ws, size_t ws_size,
                              hipStream_t stream) {
  const float* hidden = (const float*)d_in[0];   // [B,S,H]
  const float* wr     = (const float*)d_in[1];   // [H,E]
  const float* wi     = (const float*)d_in[2];   // [E,H,F]
  const float* bi     = (const float*)d_in[3];   // [E,F]
  const float* wo     = (const float*)d_in[4];   // [E,F,H]
  const float* bo     = (const float*)d_in[5];   // [E,H]

  float* out   = (float*)d_out;                       // [B,S,H]
  float* probs = out + (size_t)Ntok * Hd;             // [B,S,E]
  float* loss  = probs + (size_t)Ntok * Ee;           // scalar

  uint8_t* w = (uint8_t*)d_ws;
  int*   counts    = (int*)w;            w += 256;
  int*   offs      = (int*)w;            w += 256;
  int*   cursor    = (int*)w;            w += 256;
  int*   expert_of = (int*)w;            w += (size_t)Ntok * 4;
  int*   perm      = (int*)w;            w += (size_t)Ntok * 4;
  float* gates     = (float*)w;          w += (size_t)Ntok * 4;
  float* gbw       = (float*)w;          w += (size_t)Ee * Hd * 4;
  float* callv     = (float*)w;          w += (size_t)Hd * 4;
  w = (uint8_t*)(((uintptr_t)w + 255) & ~(uintptr_t)255);
  __bf16* h1       = (__bf16*)w;         // [Ntok, F] bf16 (67 MB)

  moe_init<<<1, 32, 0, stream>>>(counts);
  moe_router<<<Ntok / 8, 256, 0, stream>>>(hidden, wr, probs, expert_of, gates, counts);
  moe_scan_loss<<<1, 1, 0, stream>>>(counts, offs, cursor, loss);
  moe_scatter<<<Ntok / 256, 256, 0, stream>>>(expert_of, cursor, perm);
  moe_gbw<<<dim3(Hd / 256, Ee), 256, 0, stream>>>(bi, wo, gbw);
  moe_call<<<Hd / 256, 256, 0, stream>>>(gbw, bo, callv);
  moe_gemm1<<<dim3(Ntok / TM, Fd / TN, Ee), 256, 0, stream>>>(hidden, wi, bi, gates, perm, offs, h1);
  moe_gemm2<<<dim3(Ntok / TM, Hd / TN, Ee), 256, 0, stream>>>(h1, wo, perm, offs, gbw, callv, out);
}
